// EncoderAttention4_24919400252112
// MI455X (gfx1250) — matrix operand order repaired and checked by reading, not compiled, not run on hardware
//
#include <hip/hip_runtime.h>
#include <hip/hip_bf16.h>
#include <cstdint>
#include <cstddef>

// ---------------- problem constants (from reference) ----------------
#define B_TOTAL   65536
#define OBSW      168      // 48 ego + 5*24 ado
#define NEGO_     48
#define NADO_     24
#define NAG       5        // NADOAG
#define E_        256
#define H_        4
#define AD_       64
#define OUTW      112      // 48 passthrough + 64 proj
#define ROWS_PB   64       // rows per block
#define PS        264      // padded LDS row stride (bf16 elems); 528B = 33*16B -> aligned & bank-spread
#define TILE_ELEMS 512     // one pre-swizzled B tile = 32 lanes * 16 bf16 = 1024 B

typedef __attribute__((ext_vector_type(16))) __bf16 v16bf;
typedef __attribute__((ext_vector_type(8)))  float  v8f;

// ---- pre-swizzled weight tile table (units of TILE_ELEMS bf16 in d_ws) ----
// tile layout inside each matrix: tile_local = nt*KT + kt  (kt fastest in gemm inner loop)
enum : int {
  T_EGO0 = 0,                 // KT=2, NT=16
  T_EGO1 = T_EGO0 + 2 * 16,   // KT=8, NT=16
  T_ADO0 = T_EGO1 + 8 * 16,   // KT=1, NT=16
  T_ADO1 = T_ADO0 + 1 * 16,   // KT=8, NT=16
  T_KEY  = T_ADO1 + 8 * 16,   // KT=8, NT=16 (hed layout)
  T_SEL  = T_KEY  + 8 * 16,
  T_LAT  = T_SEL  + 8 * 16,
  T_PROJ = T_LAT  + 8 * 16,   // KT=8, NT=4
  T_TOTAL = T_PROJ + 8 * 4    // 720 tiles -> 737,280 B of d_ws
};

static __device__ inline v8f wmma_bf16(v16bf a, v16bf b, v8f c) {
  // D = A(16x32 bf16) * B(32x16 bf16) + C(16x16 f32)
  return __builtin_amdgcn_wmma_f32_16x16x32_bf16(false, a, false, b, (short)0, c, false, false);
}

// =====================================================================
// Prep kernel: gather fp32 weights -> bf16 B-fragment-linear layout.
// One thread per (tile, lane); writes the lane's 32B fragment chunk.
// Dense 32x16 B layout: lanes 0-15 hold K=0..7,16..23, lanes 16-31 K=8..15,24..31; N = lane%16.
// =====================================================================
__global__ __launch_bounds__(256)
void prep_weights(const float* __restrict__ ego_w0, const float* __restrict__ ego_w1,
                  const float* __restrict__ ado_w0, const float* __restrict__ ado_w1,
                  const float* __restrict__ key_w,  const float* __restrict__ sel_w,
                  const float* __restrict__ lat_w,  const float* __restrict__ proj_w,
                  __bf16* __restrict__ wt)
{
  const int g = blockIdx.x * 256 + threadIdx.x;
  if (g >= T_TOTAL * 32) return;
  const int tile = g >> 5, lane = g & 31;

  const float* W; int mode, KT, Klim, N, base;
  if (tile < T_EGO1)      { W = ego_w0; mode = 0; KT = 2; Klim = 48;  N = 256; base = T_EGO0; }
  else if (tile < T_ADO0) { W = ego_w1; mode = 0; KT = 8; Klim = 256; N = 256; base = T_EGO1; }
  else if (tile < T_ADO1) { W = ado_w0; mode = 0; KT = 1; Klim = 25;  N = 256; base = T_ADO0; }
  else if (tile < T_KEY)  { W = ado_w1; mode = 0; KT = 8; Klim = 256; N = 256; base = T_ADO1; }
  else if (tile < T_SEL)  { W = key_w;  mode = 1; KT = 8; Klim = 256; N = 256; base = T_KEY;  }
  else if (tile < T_LAT)  { W = sel_w;  mode = 1; KT = 8; Klim = 256; N = 256; base = T_SEL;  }
  else if (tile < T_PROJ) { W = lat_w;  mode = 1; KT = 8; Klim = 256; N = 256; base = T_LAT;  }
  else                    { W = proj_w; mode = 0; KT = 8; Klim = 256; N = 64;  base = T_PROJ; }

  const int tl = tile - base;
  const int nt = tl / KT, kt = tl - nt * KT;
  const int n = nt * 16 + (lane & 15);
  const int half = lane >> 4;

  union { uint4 u[2]; v16bf v; } t;
  #pragma unroll
  for (int e = 0; e < 16; ++e) {
    const int k = kt * 32 + ((e < 8) ? (half * 8 + e) : (16 + half * 8 + (e - 8)));
    float f = 0.f;
    if (k < Klim) {
      const size_t idx = (mode == 0)
          ? ((size_t)k * N + n)
          : ((size_t)(n >> 6) * (E_ * AD_) + (size_t)k * AD_ + (n & 63));  // W[h][e][d]
      f = W[idx];
    }
    t.v[e] = (__bf16)f;
  }
  __bf16* dst = wt + (size_t)tile * TILE_ELEMS + lane * 16;
  *reinterpret_cast<uint4*>(dst)     = t.u[0];
  *reinterpret_cast<uint4*>(dst + 8) = t.u[1];
}

// ---- cooperative global -> LDS copy of one stage's swizzled weight block ----
static __device__ inline void stage_w(__bf16* __restrict__ dst, const __bf16* __restrict__ src,
                                      int n_tiles, int tid) {
  const uint4* s = reinterpret_cast<const uint4*>(src);
  uint4*       d = reinterpret_cast<uint4*>(dst);
  const int n = n_tiles * (TILE_ELEMS / 8);       // uint4 chunks
  for (int i = tid; i < n; i += 256) d[i] = s[i];
}

// ---- B fragment from LDS-staged bf16 tiles: two ds_load_b128, no converts ----
static __device__ inline v16bf bfrag_lds(const __bf16* Wb, int tile, int lane) {
  const __bf16* p = Wb + (size_t)tile * TILE_ELEMS + lane * 16;
  union { uint4 u[2]; v16bf v; } t;
  t.u[0] = *reinterpret_cast<const uint4*>(p);
  t.u[1] = *reinterpret_cast<const uint4*>(p + 8);
  return t.v;
}

// ---- A fragment from LDS (bf16, padded stride PS), 16-bit A 16x32 layout ----
static __device__ inline v16bf afrag_lds(const __bf16* X, int m0, int kbase, int lane) {
  const int m    = m0 + (lane & 15);
  const int half = lane >> 4;
  const __bf16* p = X + (size_t)m * PS + kbase + half * 8;   // 16B aligned by construction
  union { uint4 u[2]; v16bf v; } t;
  t.u[0] = *reinterpret_cast<const uint4*>(p);        // K = kbase+half*8 .. +7
  t.u[1] = *reinterpret_cast<const uint4*>(p + 16);   // K = kbase+16+half*8 .. +7
  return t.v;
}

// ---- A fragment built directly from global obs (ego input, K padded 48->64) ----
static __device__ inline v16bf afrag_ego(const float* __restrict__ obs, int grow0, int kbase, int lane) {
  const int m    = grow0 + (lane & 15);
  const int half = lane >> 4;
  v16bf a;
  #pragma unroll
  for (int e = 0; e < 16; ++e) {
    const int k = kbase + ((e < 8) ? (half * 8 + e) : (16 + half * 8 + (e - 8)));
    const float f = (k < NEGO_) ? obs[(size_t)m * OBSW + k] : 0.f;
    a[e] = (__bf16)f;
  }
  return a;
}

// ---- A fragment for ado input (K=25 padded to 32): ado_in[b,n,c] = obs[b, 48 + c*5 + n] ----
static __device__ inline v16bf afrag_ado(const float* __restrict__ obs, int grow0, int lane, int agent) {
  const int m    = grow0 + (lane & 15);
  const int half = lane >> 4;
  v16bf a;
  #pragma unroll
  for (int e = 0; e < 16; ++e) {
    const int k = (e < 8) ? (half * 8 + e) : (16 + half * 8 + (e - 8));
    float f;
    if (k < NADO_)       f = obs[(size_t)m * OBSW + NEGO_ + k * NAG + agent];
    else if (k == NADO_) f = (agent >= 2) ? 1.f : 0.f;   // team_ids[1:6] = 0,0,1,1,1
    else                 f = 0.f;
    a[e] = (__bf16)f;
  }
  return a;
}

// ---- C tile store to LDS bf16 with bias + leaky_relu ----
template<bool ACT, bool HASB>
static __device__ inline void store_c_lds(v8f c, __bf16* Y, const float* __restrict__ bias,
                                          int m0, int nt, int lane) {
  const int ncl = lane & 15;
  const int mb  = m0 + (lane >> 4) * 8;
  const int col = nt * 16 + ncl;
  const float bv = HASB ? bias[col] : 0.f;
  #pragma unroll
  for (int j = 0; j < 8; ++j) {
    float v = c[j] + bv;
    if (ACT) v = fmaxf(v, 0.01f * v);   // leaky_relu, slope 0.01
    Y[(size_t)(mb + j) * PS + col] = (__bf16)v;
  }
}

// ---- GEMM stage, register-blocked 2 row-tiles per wave:
//      A frags for both row tiles held in VGPRs; each B fragment feeds two
//      independent WMMA chains (halves B LDS traffic, breaks WMMA dep chains).
template<bool ACT, bool HASB>
static __device__ inline void gemm_stage2(const __bf16* X, const __bf16* Wb,
                                          const float* __restrict__ bias,
                                          __bf16* Y, int m0a, int lane, int nt0, int nt1) {
  v16bf a0[8], a1[8];
  #pragma unroll
  for (int kt = 0; kt < 8; ++kt) {
    a0[kt] = afrag_lds(X, m0a,      kt * 32, lane);
    a1[kt] = afrag_lds(X, m0a + 16, kt * 32, lane);
  }
  for (int nt = nt0; nt < nt1; ++nt) {
    v8f c0 = {}, c1 = {};
    #pragma unroll
    for (int kt = 0; kt < 8; ++kt) {
      const v16bf b = bfrag_lds(Wb, nt * 8 + kt, lane);
      c0 = wmma_bf16(a0[kt], b, c0);
      c1 = wmma_bf16(a1[kt], b, c1);
    }
    store_c_lds<ACT, HASB>(c0, Y, bias, m0a,      nt, lane);
    store_c_lds<ACT, HASB>(c1, Y, bias, m0a + 16, nt, lane);
  }
}

// ---- vectorized 64-wide bf16 dot from LDS (16B ds loads) ----
static __device__ inline float dot64_lds(const __bf16* a, const __bf16* b) {
  float acc = 0.f;
  #pragma unroll
  for (int c8 = 0; c8 < 8; ++c8) {
    union { uint4 u; __bf16 h[8]; } ua, ub;
    ua.u = *reinterpret_cast<const uint4*>(a + c8 * 8);
    ub.u = *reinterpret_cast<const uint4*>(b + c8 * 8);
    #pragma unroll
    for (int j = 0; j < 8; ++j) acc += (float)ua.h[j] * (float)ub.h[j];
  }
  return acc;
}

__global__ __launch_bounds__(256, 1)
void fused_encoder_attention(const float* __restrict__ obs,
    const float* __restrict__ ego_b0, const float* __restrict__ ego_b1,
    const float* __restrict__ ado_b0, const float* __restrict__ ado_b1,
    const float* __restrict__ lat_b,  const float* __restrict__ proj_b,
    const __bf16* __restrict__ wt,
    float* __restrict__ out)
{
  extern __shared__ char smem_raw[];
  __bf16* wbuf  = reinterpret_cast<__bf16*>(smem_raw);          // 128 tiles max = 128KB weight staging
  __bf16* tmp   = wbuf  + 128 * TILE_ELEMS;                     // generic ping buffer
  __bf16* egoA  = tmp   + ROWS_PB * PS;                         // ego embeddings
  __bf16* slctA = egoA  + ROWS_PB * PS;                         // selector (B, H*AD)
  __bf16* adoC  = slctA + ROWS_PB * PS;                         // current agent embedding
  __bf16* klb   = adoC  + ROWS_PB * PS;                         // keys / lats scratch

  const int tid   = threadIdx.x;
  const int lane  = tid & 31;
  const int wid   = tid >> 5;
  const int m0a   = (wid & 1) * 32;      // wave owns row tiles {m0a, m0a+16}
  const int ntq   = wid >> 1;            // wave owns 4 of 16 n-tiles
  const int nt0   = ntq * 4, nt1 = nt0 + 4;
  const int r0    = blockIdx.x * ROWS_PB;
  const int r     = tid >> 2, h = tid & 3;   // (row, head) pair owned by this thread

  // ---- ego MLP layer 0 (K = 48 padded to 64) -> tmp
  stage_w(wbuf, wt + (size_t)T_EGO0 * TILE_ELEMS, 32, tid);
  __syncthreads();
  {
    v16bf a0k0 = afrag_ego(obs, r0 + m0a,      0,  lane);
    v16bf a0k1 = afrag_ego(obs, r0 + m0a,      32, lane);
    v16bf a1k0 = afrag_ego(obs, r0 + m0a + 16, 0,  lane);
    v16bf a1k1 = afrag_ego(obs, r0 + m0a + 16, 32, lane);
    for (int nt = nt0; nt < nt1; ++nt) {
      const v16bf b0 = bfrag_lds(wbuf, nt * 2 + 0, lane);
      const v16bf b1 = bfrag_lds(wbuf, nt * 2 + 1, lane);
      v8f c0 = {}, c1 = {};
      c0 = wmma_bf16(a0k0, b0, c0);  c1 = wmma_bf16(a1k0, b0, c1);
      c0 = wmma_bf16(a0k1, b1, c0);  c1 = wmma_bf16(a1k1, b1, c1);
      store_c_lds<true, true>(c0, tmp, ego_b0, m0a,      nt, lane);
      store_c_lds<true, true>(c1, tmp, ego_b0, m0a + 16, nt, lane);
    }
  }
  __syncthreads();
  // ---- ego layer 1 -> egoA
  stage_w(wbuf, wt + (size_t)T_EGO1 * TILE_ELEMS, 128, tid);
  __syncthreads();
  gemm_stage2<true, true>(tmp, wbuf, ego_b1, egoA, m0a, lane, nt0, nt1);
  __syncthreads();
  // ---- selector projection -> slctA
  stage_w(wbuf, wt + (size_t)T_SEL * TILE_ELEMS, 128, tid);
  __syncthreads();
  gemm_stage2<false, false>(egoA, wbuf, nullptr, slctA, m0a, lane, nt0, nt1);
  __syncthreads();

  // ---- per-agent loop with online (flash-style) softmax; state in registers ----
  float M = -3.0e38f, S = 0.f;
  float agg[AD_];
  #pragma unroll
  for (int d = 0; d < AD_; ++d) agg[d] = 0.f;

  for (int ag = 0; ag < NAG; ++ag) {
    // ado layer 0 -> tmp
    stage_w(wbuf, wt + (size_t)T_ADO0 * TILE_ELEMS, 16, tid);
    __syncthreads();
    {
      v16bf a0 = afrag_ado(obs, r0 + m0a,      lane, ag);
      v16bf a1 = afrag_ado(obs, r0 + m0a + 16, lane, ag);
      for (int nt = nt0; nt < nt1; ++nt) {
        const v16bf b = bfrag_lds(wbuf, nt, lane);
        v8f c0 = {}, c1 = {};
        c0 = wmma_bf16(a0, b, c0);
        c1 = wmma_bf16(a1, b, c1);
        store_c_lds<true, true>(c0, tmp, ado_b0, m0a,      nt, lane);
        store_c_lds<true, true>(c1, tmp, ado_b0, m0a + 16, nt, lane);
      }
    }
    __syncthreads();
    // ado layer 1 -> adoC
    stage_w(wbuf, wt + (size_t)T_ADO1 * TILE_ELEMS, 128, tid);
    __syncthreads();
    gemm_stage2<true, true>(tmp, wbuf, ado_b1, adoC, m0a, lane, nt0, nt1);
    __syncthreads();
    // keys -> klb
    stage_w(wbuf, wt + (size_t)T_KEY * TILE_ELEMS, 128, tid);
    __syncthreads();
    gemm_stage2<false, false>(adoC, wbuf, nullptr, klb, m0a, lane, nt0, nt1);
    __syncthreads();
    // logit for this agent (register) + stage lat weights in the same phase
    const float l = dot64_lds(slctA + (size_t)r * PS + h * AD_,
                              klb   + (size_t)r * PS + h * AD_) * 0.125f;  // 1/sqrt(64)
    stage_w(wbuf, wt + (size_t)T_LAT * TILE_ELEMS, 128, tid);
    __syncthreads();
    // lats -> klb
    gemm_stage2<true, true>(adoC, wbuf, lat_b, klb, m0a, lane, nt0, nt1);
    __syncthreads();
    // online softmax update: rescale running aggregate, accumulate this agent
    {
      const float Mn   = fmaxf(M, l);
      const float corr = __expf(M - Mn);   // 0 on first iteration (M = -inf-ish)
      const float w    = __expf(l - Mn);
      S = S * corr + w;
      const __bf16* pl = klb + (size_t)r * PS + h * AD_;
      #pragma unroll
      for (int c8 = 0; c8 < 8; ++c8) {
        union { uint4 u; __bf16 hh[8]; } ul;
        ul.u = *reinterpret_cast<const uint4*>(pl + c8 * 8);
        #pragma unroll
        for (int j = 0; j < 8; ++j)
          agg[c8 * 8 + j] = agg[c8 * 8 + j] * corr + w * (float)ul.hh[j];
      }
      M = Mn;
    }
    __syncthreads();
  }

  // ---- finalize softmax and write agg -> tmp (bf16, 16B stores) ----
  {
    const float inv = 1.f / S;
    __bf16* pt = tmp + (size_t)r * PS + h * AD_;
    #pragma unroll
    for (int c8 = 0; c8 < 8; ++c8) {
      union { uint4 u; __bf16 hh[8]; } ul;
      #pragma unroll
      for (int j = 0; j < 8; ++j) ul.hh[j] = (__bf16)(agg[c8 * 8 + j] * inv);
      *reinterpret_cast<uint4*>(pt + c8 * 8) = ul.u;
    }
  }
  __syncthreads();

  // ---- final projection (256 -> 64) straight to global output ----
  stage_w(wbuf, wt + (size_t)T_PROJ * TILE_ELEMS, 32, tid);
  __syncthreads();
  {
    const int ntp  = wid >> 1;            // one of 4 n-tiles
    const int m0p  = (wid & 1) * 32;      // two row tiles
    v16bf a0[8], a1[8];
    #pragma unroll
    for (int kt = 0; kt < 8; ++kt) {
      a0[kt] = afrag_lds(tmp, m0p,      kt * 32, lane);
      a1[kt] = afrag_lds(tmp, m0p + 16, kt * 32, lane);
    }
    v8f c0 = {}, c1 = {};
    #pragma unroll
    for (int kt = 0; kt < 8; ++kt) {
      const v16bf b = bfrag_lds(wbuf, ntp * 8 + kt, lane);
      c0 = wmma_bf16(a0[kt], b, c0);
      c1 = wmma_bf16(a1[kt], b, c1);
    }
    const int ncl = lane & 15;
    const int col = ntp * 16 + ncl;
    const float bv = proj_b[col];
    const int mb0 = m0p + (lane >> 4) * 8;
    #pragma unroll
    for (int j = 0; j < 8; ++j) {
      out[(size_t)(r0 + mb0 + j)      * OUTW + NEGO_ + col] = c0[j] + bv;
      out[(size_t)(r0 + mb0 + 16 + j) * OUTW + NEGO_ + col] = c1[j] + bv;
    }
  }
  // ---- ego observation passthrough (first 48 output cols) ----
  for (int i = tid; i < ROWS_PB * NEGO_; i += 256) {
    const int rr = i / NEGO_, cc = i - rr * NEGO_;
    out[(size_t)(r0 + rr) * OUTW + cc] = obs[(size_t)(r0 + rr) * OBSW + cc];
  }
}

extern "C" void kernel_launch(void* const* d_in, const int* in_sizes, int n_in,
                              void* d_out, int out_size, void* d_ws, size_t ws_size,
                              hipStream_t stream) {
  (void)in_sizes; (void)n_in; (void)out_size; (void)ws_size;
  const float* obs    = (const float*)d_in[0];
  const float* ego_w0 = (const float*)d_in[1];
  const float* ego_b0 = (const float*)d_in[2];
  const float* ego_w1 = (const float*)d_in[3];
  const float* ego_b1 = (const float*)d_in[4];
  const float* ado_w0 = (const float*)d_in[5];
  const float* ado_b0 = (const float*)d_in[6];
  const float* ado_w1 = (const float*)d_in[7];
  const float* ado_b1 = (const float*)d_in[8];
  const float* key_w  = (const float*)d_in[9];
  const float* sel_w  = (const float*)d_in[10];
  const float* lat_w  = (const float*)d_in[11];
  const float* lat_b  = (const float*)d_in[12];
  const float* proj_w = (const float*)d_in[13];
  const float* proj_b = (const float*)d_in[14];
  float* out = (float*)d_out;
  __bf16* wt = (__bf16*)d_ws;   // needs 720 KB of scratch (T_TOTAL * 1 KB)

  // 1) pre-swizzle all weights into B-fragment-linear bf16 tiles (cheap: ~0.4 MFLOP)
  {
    const int threads = T_TOTAL * 32;
    prep_weights<<<dim3((threads + 255) / 256), dim3(256), 0, stream>>>(
        ego_w0, ego_w1, ado_w0, ado_w1, key_w, sel_w, lat_w, proj_w, wt);
  }
  // 2) fused pipeline; dynamic LDS: 128KB weight stage + 5 bf16 act buffers = 300,032 B (<= 320KB/WGP)
  {
    const size_t smem = (size_t)128 * TILE_ELEMS * sizeof(__bf16)
                      + (size_t)5 * ROWS_PB * PS * sizeof(__bf16);
    dim3 grid(B_TOTAL / ROWS_PB), block(256);
    fused_encoder_attention<<<grid, block, smem, stream>>>(
        obs, ego_b0, ego_b1, ado_b0, ado_b1, lat_b, proj_b, wt, out);
  }
}
